// AntiviralGNN_3607772529402
// MI455X (gfx1250) — matrix operand order repaired and checked by reading, not compiled, not run on hardware
//
#include <hip/hip_runtime.h>
#include <hip/hip_bf16.h>
#include <math.h>

// Problem constants (from reference)
#define Nn 100000
#define Ee 400000
#define Bb 4096
#define FAc 30
#define FBc 11
#define Hh 256
#define H2 512
#define Ll 5

typedef __attribute__((ext_vector_type(16))) __bf16 bf16x16;
typedef __attribute__((ext_vector_type(8)))  __bf16 bf16x8;
typedef __attribute__((ext_vector_type(8)))  float  f32x8;

// ---------------- WMMA fragment loaders ----------------

// A fragment, 16x32 bf16, row-major source with leading dim ld.
// lanes 0-15: M=lane, elems 0..7 = K k0+0..7, elems 8..15 = K k0+16..23
// lanes 16-31: M=lane-16, K base shifted by +8 / +24.
__device__ __forceinline__ bf16x16 frag_a(const __bf16* base, int ld, int k0, int lane) {
    int m  = lane & 15;
    int kb = k0 + ((lane >> 4) << 3);
    const __bf16* p = base + (long)m * ld + kb;
    bf16x8 lo = *(const bf16x8*)p;
    bf16x8 hi = *(const bf16x8*)(p + 16);
    bf16x16 r;
#pragma unroll
    for (int j = 0; j < 8; ++j) { r[j] = lo[j]; r[8 + j] = hi[j]; }
    return r;
}

// B fragment, 32x16 bf16 from pre-transposed weights Wt[N][K] (ld = K dim).
// lanes 0-15: K = k0+0..7, k0+16..23 ; lanes 16-31: K = k0+8..15, k0+24..31 ; N = n0 + (lane&15)
__device__ __forceinline__ bf16x16 frag_b(const __bf16* Wt, int ld, int n0, int k0, int lane) {
    int nn = lane & 15;
    int kb = k0 + ((lane >> 4) << 3);
    const __bf16* p = Wt + (long)(n0 + nn) * ld + kb;
    bf16x8 lo = *(const bf16x8*)p;
    bf16x8 hi = *(const bf16x8*)(p + 16);
    bf16x16 r;
#pragma unroll
    for (int j = 0; j < 8; ++j) { r[j] = lo[j]; r[8 + j] = hi[j]; }
    return r;
}

// ---------------- utility kernels ----------------

__global__ __launch_bounds__(256) void k_zero_f32(float* p, long n) {
    long i = (long)blockIdx.x * 256 + threadIdx.x;
    if (i < n) p[i] = 0.f;
}

__global__ __launch_bounds__(256) void k_zero_i32(int* p, long n) {
    long i = (long)blockIdx.x * 256 + threadIdx.x;
    if (i < n) p[i] = 0;
}

// W1[L][K=256][N=512] f32  ->  W1t[L][N=512][K=256] bf16
__global__ __launch_bounds__(256) void k_conv_w1(const float* __restrict__ W1, __bf16* __restrict__ W1t) {
    long i = (long)blockIdx.x * 256 + threadIdx.x;
    if (i >= (long)Ll * H2 * Hh) return;
    int l = (int)(i / (H2 * Hh));
    int r = (int)(i % (H2 * Hh));
    int n = r / Hh, k = r % Hh;
    W1t[i] = (__bf16)W1[((long)l * Hh + k) * H2 + n];
}

// W2[L][K=512][N=256] f32  ->  W2t[L][N=256][K=512] bf16
__global__ __launch_bounds__(256) void k_conv_w2(const float* __restrict__ W2, __bf16* __restrict__ W2t) {
    long i = (long)blockIdx.x * 256 + threadIdx.x;
    if (i >= (long)Ll * Hh * H2) return;
    int l = (int)(i / (Hh * H2));
    int r = (int)(i % (Hh * H2));
    int n = r / H2, k = r % H2;
    W2t[i] = (__bf16)W2[((long)l * H2 + k) * Hh + n];
}

// h = x @ atom_W + atom_b   (K=30, memory bound)
__global__ __launch_bounds__(256) void k_atom(const float* __restrict__ x, const float* __restrict__ W,
                                              const float* __restrict__ b, float* __restrict__ h) {
    long idx = (long)blockIdx.x * 256 + threadIdx.x;   // exactly N*H threads
    long n = idx >> 8; int c = (int)(idx & 255);
    float s = b[c];
#pragma unroll
    for (int k = 0; k < FAc; ++k) s += x[n * FAc + k] * W[k * Hh + c];
    h[idx] = s;
}

// e = edge_attr @ bond_W + bond_b  (K=11)
__global__ __launch_bounds__(256) void k_bond(const float* __restrict__ ea, const float* __restrict__ W,
                                              const float* __restrict__ b, float* __restrict__ e) {
    long idx = (long)blockIdx.x * 256 + threadIdx.x;   // exactly E*H threads
    long n = idx >> 8; int c = (int)(idx & 255);
    float s = b[c];
#pragma unroll
    for (int k = 0; k < FBc; ++k) s += ea[n * FBc + k] * W[k * Hh + c];
    e[idx] = s;
}

// msg = relu(h[src]+e); agg[dst] += msg  (atomic scatter; skip zeros post-ReLU)
__global__ __launch_bounds__(256) void k_msg(const float* __restrict__ h, const float* __restrict__ e,
                                             const int* __restrict__ src, const int* __restrict__ dst,
                                             float* __restrict__ agg) {
    long idx = (long)blockIdx.x * 256 + threadIdx.x;   // exactly E*H threads
    long ed = idx >> 8; int c = (int)(idx & 255);
    int s = src[ed], d = dst[ed];
    float m = h[(long)s * Hh + c] + e[idx];
    m = fmaxf(m, 0.f);
    if (m != 0.f) atomicAdd(&agg[(long)d * Hh + c], m);
}

// ---------------- fused GINE node MLP (WMMA bf16) ----------------
// 32 nodes / block, 8 waves = 2 M-tiles x 4 N-wave-groups.
// A fragments register-cached; B streamed from L2-resident bf16 weights.
__global__ __launch_bounds__(256) void k_node_mlp(
    float* __restrict__ h, const float* __restrict__ agg,
    const __bf16* __restrict__ W1t, const float* __restrict__ b1,
    const float* __restrict__ g1, const float* __restrict__ be1,
    const float* __restrict__ m1, const float* __restrict__ v1,
    const __bf16* __restrict__ W2t, const float* __restrict__ b2,
    const float* __restrict__ g2, const float* __restrict__ be2,
    const float* __restrict__ m2, const float* __restrict__ v2,
    const float* __restrict__ eps, int layer)
{
    __shared__ __align__(16) __bf16 zA[32 * Hh];    // 16 KB: MLP input tile (bf16)
    __shared__ __align__(16) __bf16 tA[32 * H2];    // 32 KB: intermediate tile (bf16)
    __shared__ float hres[32 * Hh];                 // 32 KB: f32 residual
    int tid = threadIdx.x, lane = tid & 31, wv = tid >> 5;
    int mw = wv >> 2;                               // 0..1: which 16-row M-tile
    int nw = wv & 3;                                // 0..3: N-tile wave group
    long row0 = (long)blockIdx.x * 32;              // N divisible by 32
    float epsn = 1.0f + eps[layer];

    for (int i = tid; i < 32 * Hh; i += 256) {
        int r = i >> 8, c = i & 255;
        long gi = (row0 + r) * Hh + c;
        float hv = h[gi], av = agg[gi];
        hres[i] = hv;
        zA[i] = (__bf16)(epsn * hv + av);
    }
    __syncthreads();

    // GEMM1: [32 x 256] x [256 x 512]
    {
        const __bf16* zrow = zA + mw * 16 * Hh;
        bf16x16 af[8];
#pragma unroll
        for (int kk = 0; kk < 8; ++kk) af[kk] = frag_a(zrow, Hh, kk * 32, lane);
        for (int nt = nw; nt < H2 / 16; nt += 4) {
            int n0 = nt * 16;
            f32x8 acc = {0.f, 0.f, 0.f, 0.f, 0.f, 0.f, 0.f, 0.f};
#pragma unroll
            for (int kk = 0; kk < 8; ++kk) {
                bf16x16 b = frag_b(W1t, Hh, n0, kk * 32, lane);
                acc = __builtin_amdgcn_wmma_f32_16x16x32_bf16(false, af[kk], false, b, (short)0, acc, false, false);
            }
            int col = n0 + (lane & 15);
            int mb  = mw * 16 + ((lane >> 4) << 3);
            float sc = g1[col] * rsqrtf(v1[col] + 1e-5f);
            float bia = b1[col], mu = m1[col], bo = be1[col];
#pragma unroll
            for (int v = 0; v < 8; ++v) {
                float val = (acc[v] + bia - mu) * sc + bo;
                tA[(mb + v) * H2 + col] = (__bf16)fmaxf(val, 0.f);
            }
        }
    }
    __syncthreads();

    // GEMM2: [32 x 512] x [512 x 256]
    {
        const __bf16* trow = tA + mw * 16 * H2;
        bf16x16 af[16];
#pragma unroll
        for (int kk = 0; kk < 16; ++kk) af[kk] = frag_a(trow, H2, kk * 32, lane);
        for (int nt = nw; nt < Hh / 16; nt += 4) {
            int n0 = nt * 16;
            f32x8 acc = {0.f, 0.f, 0.f, 0.f, 0.f, 0.f, 0.f, 0.f};
#pragma unroll
            for (int kk = 0; kk < 16; ++kk) {
                bf16x16 b = frag_b(W2t, H2, n0, kk * 32, lane);
                acc = __builtin_amdgcn_wmma_f32_16x16x32_bf16(false, af[kk], false, b, (short)0, acc, false, false);
            }
            int col = n0 + (lane & 15);
            int mb  = mw * 16 + ((lane >> 4) << 3);
            float sc = g2[col] * rsqrtf(v2[col] + 1e-5f);
            float bia = b2[col], mu = m2[col], bo = be2[col];
#pragma unroll
            for (int v = 0; v < 8; ++v) {
                float val = (acc[v] + bia - mu) * sc + bo;
                val = fmaxf(val, 0.f) + hres[(mb + v) * Hh + col];
                h[(row0 + mb + v) * Hh + col] = val;
            }
        }
    }
}

// ---------------- Set2Set ----------------

// batch is sorted: record [start,end) per graph. Buffers pre-zeroed (empty -> [0,0)).
__global__ __launch_bounds__(256) void k_seg_bounds(const int* __restrict__ batch,
                                                    int* __restrict__ gstart, int* __restrict__ gend) {
    int n = blockIdx.x * 256 + threadIdx.x;
    if (n >= Nn) return;
    int b = batch[n];
    if (n == 0 || batch[n - 1] != b) gstart[b] = n;
    if (n == Nn - 1 || batch[n + 1] != b) gend[b] = n + 1;
}

__global__ __launch_bounds__(256) void k_lstm(const float* __restrict__ q_star, float* __restrict__ hst,
                                              float* __restrict__ cst,
                                              const float* __restrict__ Wih, const float* __restrict__ Whh,
                                              const float* __restrict__ bih, const float* __restrict__ bhh) {
    __shared__ float qs[H2];
    __shared__ float hs[Hh];
    int b = blockIdx.x, u = threadIdx.x;
    qs[u]        = q_star[(long)b * H2 + u];
    qs[u + 256]  = q_star[(long)b * H2 + u + 256];
    hs[u]        = hst[(long)b * Hh + u];
    __syncthreads();
    float gi = bih[u] + bhh[u];
    float gf = bih[Hh + u] + bhh[Hh + u];
    float gg = bih[2 * Hh + u] + bhh[2 * Hh + u];
    float go = bih[3 * Hh + u] + bhh[3 * Hh + u];
    const float* wi = Wih + (long)u * H2;
    const float* wf = Wih + (long)(Hh + u) * H2;
    const float* wg = Wih + (long)(2 * Hh + u) * H2;
    const float* wo = Wih + (long)(3 * Hh + u) * H2;
    for (int k = 0; k < H2; ++k) {
        float q = qs[k];
        gi += q * wi[k]; gf += q * wf[k]; gg += q * wg[k]; go += q * wo[k];
    }
    const float* hi = Whh + (long)u * Hh;
    const float* hf = Whh + (long)(Hh + u) * Hh;
    const float* hg = Whh + (long)(2 * Hh + u) * Hh;
    const float* ho = Whh + (long)(3 * Hh + u) * Hh;
    for (int k = 0; k < Hh; ++k) {
        float hv = hs[k];
        gi += hv * hi[k]; gf += hv * hf[k]; gg += hv * hg[k]; go += hv * ho[k];
    }
    float igv = 1.f / (1.f + expf(-gi));
    float fgv = 1.f / (1.f + expf(-gf));
    float ogv = 1.f / (1.f + expf(-go));
    float ggv = tanhf(gg);
    float c = fgv * cst[(long)b * Hh + u] + igv * ggv;
    cst[(long)b * Hh + u] = c;
    hst[(long)b * Hh + u] = ogv * tanhf(c);
}

// q_star[:, :H] = hst  (r half overwritten by k_attn)
__global__ __launch_bounds__(256) void k_qprep(float* __restrict__ q_star, const float* __restrict__ hst) {
    long i = (long)blockIdx.x * 256 + threadIdx.x;   // B*H threads
    long b = i >> 8; int u = (int)(i & 255);
    q_star[b * H2 + u] = hst[i];
}

// Fused attention: block per graph, contiguous node segment, no global atomics.
// phase1: ener[n]=h[n,:].q (wave/node) + block max
// phase2: a=exp(ener-emax), block sum
// phase3: r[c]=sum_n a[n]*h[n,c]/asum  (coalesced column-per-thread)
__global__ __launch_bounds__(256) void k_attn(const float* __restrict__ h, const float* __restrict__ hst,
                                              const int* __restrict__ gstart, const int* __restrict__ gend,
                                              float* __restrict__ aatt, float* __restrict__ q_star) {
    __shared__ float q[Hh];
    __shared__ float red[8];
    __shared__ float s_emax, s_asum;
    int b = blockIdx.x, tid = threadIdx.x, lane = tid & 31, wv = tid >> 5;
    int s0 = gstart[b], s1 = gend[b];
    q[tid] = hst[(long)b * Hh + tid];
    __syncthreads();

    // phase 1: energies + max
    float wmax = -3.4e38f;
    for (int n = s0 + wv; n < s1; n += 8) {
        float s = 0.f;
#pragma unroll
        for (int j = 0; j < 8; ++j) {
            int c = lane + j * 32;
            s += h[(long)n * Hh + c] * q[c];
        }
#pragma unroll
        for (int m = 16; m > 0; m >>= 1) s += __shfl_xor(s, m, 32);
        wmax = fmaxf(wmax, s);              // butterfly leaves full sum in all lanes
        if (lane == 0) aatt[n] = s;         // stash raw energy
    }
    if (lane == 0) red[wv] = wmax;
    __syncthreads();
    if (tid == 0) {
        float m = red[0];
#pragma unroll
        for (int i = 1; i < 8; ++i) m = fmaxf(m, red[i]);
        s_emax = (s1 > s0) ? m : 0.f;       // empty segment -> 0 (isfinite fallback)
    }
    __syncthreads();

    // phase 2: exp + sum
    float part = 0.f;
    for (int n = s0 + tid; n < s1; n += 256) {
        float av = expf(aatt[n] - s_emax);
        aatt[n] = av;
        part += av;
    }
#pragma unroll
    for (int m = 16; m > 0; m >>= 1) part += __shfl_xor(part, m, 32);
    if (lane == 0) red[wv] = part;
    __syncthreads();
    if (tid == 0) {
        float s = 0.f;
#pragma unroll
        for (int i = 0; i < 8; ++i) s += red[i];
        s_asum = s + 1e-16f;
    }
    __syncthreads();

    // phase 3: weighted sum, thread owns one column (fully coalesced over nodes)
    float r = 0.f;
    for (int n = s0; n < s1; ++n) r += aatt[n] * h[(long)n * Hh + tid];
    q_star[(long)b * H2 + Hh + tid] = r / s_asum;
}

// ---------------- classifier ----------------

__global__ __launch_bounds__(256) void k_fc1(const float* __restrict__ q_star, const float* __restrict__ W,
                                             const float* __restrict__ bia, const float* __restrict__ g,
                                             const float* __restrict__ be, const float* __restrict__ mu,
                                             const float* __restrict__ var, float* __restrict__ o1) {
    __shared__ float qs[H2];
    int b = blockIdx.x, u = threadIdx.x;
    qs[u]       = q_star[(long)b * H2 + u];
    qs[u + 256] = q_star[(long)b * H2 + u + 256];
    __syncthreads();
    float s = bia[u];
    for (int k = 0; k < H2; ++k) s += qs[k] * W[(long)k * Hh + u];
    s = (s - mu[u]) * rsqrtf(var[u] + 1e-5f) * g[u] + be[u];
    o1[(long)b * Hh + u] = fmaxf(s, 0.f);
}

__global__ __launch_bounds__(128) void k_fc2(const float* __restrict__ o1, const float* __restrict__ W,
                                             const float* __restrict__ bia, const float* __restrict__ g,
                                             const float* __restrict__ be, const float* __restrict__ mu,
                                             const float* __restrict__ var, float* __restrict__ o2) {
    __shared__ float r[Hh];
    int b = blockIdx.x, u = threadIdx.x;
    r[u]       = o1[(long)b * Hh + u];
    r[u + 128] = o1[(long)b * Hh + u + 128];
    __syncthreads();
    float s = bia[u];
    for (int k = 0; k < Hh; ++k) s += r[k] * W[k * 128 + u];
    s = (s - mu[u]) * rsqrtf(var[u] + 1e-5f) * g[u] + be[u];
    o2[(long)b * 128 + u] = fmaxf(s, 0.f);
}

__global__ __launch_bounds__(256) void k_fc3(const float* __restrict__ o2, const float* __restrict__ W,
                                             const float* __restrict__ bia, float* __restrict__ out) {
    int b = blockIdx.x * 256 + threadIdx.x;
    if (b >= Bb) return;
    float s = bia[0];
    for (int k = 0; k < 128; ++k) s += o2[(long)b * 128 + k] * W[k];
    out[b] = s;
}

// ---------------- host launch ----------------

static inline size_t al256(size_t x) { return (x + 255) & ~(size_t)255; }

extern "C" void kernel_launch(void* const* d_in, const int* in_sizes, int n_in,
                              void* d_out, int out_size, void* d_ws, size_t ws_size,
                              hipStream_t stream) {
    const float* x         = (const float*)d_in[0];
    const float* edge_attr = (const float*)d_in[1];
    const int*   edge_idx  = (const int*)d_in[2];
    const int*   batch     = (const int*)d_in[3];
    const float* atom_W    = (const float*)d_in[4];
    const float* atom_b    = (const float*)d_in[5];
    const float* bond_W    = (const float*)d_in[6];
    const float* bond_b    = (const float*)d_in[7];
    const float* eps       = (const float*)d_in[8];
    const float* W1        = (const float*)d_in[9];
    const float* b1        = (const float*)d_in[10];
    const float* bn1_g     = (const float*)d_in[11];
    const float* bn1_b     = (const float*)d_in[12];
    const float* bn1_m     = (const float*)d_in[13];
    const float* bn1_v     = (const float*)d_in[14];
    const float* W2        = (const float*)d_in[15];
    const float* b2        = (const float*)d_in[16];
    const float* bn_g      = (const float*)d_in[17];
    const float* bn_b      = (const float*)d_in[18];
    const float* bn_m      = (const float*)d_in[19];
    const float* bn_v      = (const float*)d_in[20];
    const float* Wih       = (const float*)d_in[21];
    const float* Whh       = (const float*)d_in[22];
    const float* bih       = (const float*)d_in[23];
    const float* bhh       = (const float*)d_in[24];
    const float* cW1       = (const float*)d_in[25];
    const float* cb1       = (const float*)d_in[26];
    const float* cbn1_g    = (const float*)d_in[27];
    const float* cbn1_b    = (const float*)d_in[28];
    const float* cbn1_m    = (const float*)d_in[29];
    const float* cbn1_v    = (const float*)d_in[30];
    const float* cW2       = (const float*)d_in[31];
    const float* cb2       = (const float*)d_in[32];
    const float* cbn2_g    = (const float*)d_in[33];
    const float* cbn2_b    = (const float*)d_in[34];
    const float* cbn2_m    = (const float*)d_in[35];
    const float* cbn2_v    = (const float*)d_in[36];
    const float* cW3       = (const float*)d_in[37];
    const float* cb3       = (const float*)d_in[38];

    const int* src = edge_idx;
    const int* dst = edge_idx + Ee;

    char* ws = (char*)d_ws;
    size_t off = 0;
    float*  h      = (float*)(ws + off);  off += al256((size_t)Nn * Hh * 4);
    float*  e      = (float*)(ws + off);  off += al256((size_t)Ee * Hh * 4);
    float*  agg    = (float*)(ws + off);  off += al256((size_t)Nn * Hh * 4);
    __bf16* W1t    = (__bf16*)(ws + off); off += al256((size_t)Ll * H2 * Hh * 2);
    __bf16* W2t    = (__bf16*)(ws + off); off += al256((size_t)Ll * Hh * H2 * 2);
    float*  q_star = (float*)(ws + off);  off += al256((size_t)Bb * H2 * 4);
    float*  hst    = (float*)(ws + off);  off += al256((size_t)Bb * Hh * 4);
    float*  cst    = (float*)(ws + off);  off += al256((size_t)Bb * Hh * 4);
    float*  aatt   = (float*)(ws + off);  off += al256((size_t)Nn * 4);
    int*    gstart = (int*)(ws + off);    off += al256((size_t)Bb * 4);
    int*    gend   = (int*)(ws + off);    off += al256((size_t)Bb * 4);
    float*  o1     = (float*)(ws + off);  off += al256((size_t)Bb * Hh * 4);
    float*  o2     = (float*)(ws + off);  off += al256((size_t)Bb * 128 * 4);
    (void)ws_size; (void)in_sizes; (void)n_in; (void)out_size;

    // weights -> bf16 transposed [N][K]
    k_conv_w1<<<(Ll * H2 * Hh + 255) / 256, 256, 0, stream>>>(W1, W1t);
    k_conv_w2<<<(Ll * Hh * H2 + 255) / 256, 256, 0, stream>>>(W2, W2t);

    // embeddings
    k_atom<<<(Nn * Hh) / 256, 256, 0, stream>>>(x, atom_W, atom_b, h);
    k_bond<<<((long)Ee * Hh) / 256, 256, 0, stream>>>(edge_attr, bond_W, bond_b, e);

    // graph segment bounds (batch sorted); zero first so empty graphs -> [0,0)
    k_zero_i32<<<(2 * Bb + 255) / 256, 256, 0, stream>>>(gstart, 2L * Bb);  // gstart+gend contiguous
    k_seg_bounds<<<(Nn + 255) / 256, 256, 0, stream>>>(batch, gstart, gend);

    // GINE layers
    for (int l = 0; l < Ll; ++l) {
        k_zero_f32<<<((long)Nn * Hh + 255) / 256, 256, 0, stream>>>(agg, (long)Nn * Hh);
        k_msg<<<((long)Ee * Hh) / 256, 256, 0, stream>>>(h, e, src, dst, agg);
        k_node_mlp<<<Nn / 32, 256, 0, stream>>>(
            h, agg,
            W1t + (size_t)l * H2 * Hh, b1 + (size_t)l * H2,
            bn1_g + (size_t)l * H2, bn1_b + (size_t)l * H2,
            bn1_m + (size_t)l * H2, bn1_v + (size_t)l * H2,
            W2t + (size_t)l * Hh * H2, b2 + (size_t)l * Hh,
            bn_g + (size_t)l * Hh, bn_b + (size_t)l * Hh,
            bn_m + (size_t)l * Hh, bn_v + (size_t)l * Hh,
            eps, l);
    }

    // Set2Set state init (q_star, hst, cst contiguous -> one zero pass)
    k_zero_f32<<<((long)Bb * (H2 + 2 * Hh) + 255) / 256, 256, 0, stream>>>(q_star, (long)Bb * (H2 + 2 * Hh));

    for (int s = 0; s < 3; ++s) {
        k_lstm<<<Bb, 256, 0, stream>>>(q_star, hst, cst, Wih, Whh, bih, bhh);
        k_qprep<<<((long)Bb * Hh) / 256, 256, 0, stream>>>(q_star, hst);
        k_attn<<<Bb, 256, 0, stream>>>(h, hst, gstart, gend, aatt, q_star);
    }

    // classifier
    k_fc1<<<Bb, 256, 0, stream>>>(q_star, cW1, cb1, cbn1_g, cbn1_b, cbn1_m, cbn1_v, o1);
    k_fc2<<<Bb, 128, 0, stream>>>(o1, cW2, cb2, cbn2_g, cbn2_b, cbn2_m, cbn2_v, o2);
    k_fc3<<<(Bb + 255) / 256, 256, 0, stream>>>(o2, cW3, cb3, (float*)d_out);
}
